// TTCN_34797825032291
// MI455X (gfx1250) — matrix operand order repaired and checked by reading, not compiled, not run on hardware
//
#include <hip/hip_runtime.h>

// ---- CDNA5 WMMA vector types
typedef __attribute__((ext_vector_type(16))) _Float16 v16h;
typedef __attribute__((ext_vector_type(8)))  _Float16 v8h;
typedef __attribute__((ext_vector_type(8)))  float    v8f;

#define D_IN   32      // INPUT_DIM
#define K_DIM  63      // TTCN_DIM
#define HID    64      // padded hidden (63 -> 64, pad col == 0)
#define LXN    512
#define NBATCH 128
#define COLS   2016    // 32*63
#define L2E    1.4426950408889634f
#define NEG_BIG (-1.0e8f)

__device__ __forceinline__ v16h cat8(v8h lo, v8h hi) {
    return __builtin_shufflevector(lo, hi, 0,1,2,3,4,5,6,7,8,9,10,11,12,13,14,15);
}

// One workgroup = one (n, k-split). split 0 -> k 0..31, split 1 -> k 32..62.
// 256 threads = 8 wave32s. Fully fused: MLP (WMMA) -> online masked softmax
// over Lx -> X-weighted pooling, no global intermediates.
__launch_bounds__(256, 1)
__global__ void ttcn_fused(const float* __restrict__ pX,    // (128,512,32)
                           const float* __restrict__ pMask, // (128,512,1)
                           const float* __restrict__ pW1,   // (32,63)
                           const float* __restrict__ pB1,   // (63)
                           const float* __restrict__ pW2,   // (63,63)
                           const float* __restrict__ pB2,   // (63)
                           const float* __restrict__ pW3,   // (63,2016)
                           const float* __restrict__ pB3,   // (2016)
                           const float* __restrict__ pTB,   // (1,63)
                           float* __restrict__ pOut)        // (128,63)
{
    __shared__ _Float16 sX [LXN * D_IN];   // 32 KB  X[n] as f16
    __shared__ _Float16 sH1[LXN * HID];    // 64 KB  relu(X@W1+b1), col 63 = 0
    __shared__ _Float16 sH2[LXN * HID];    // 64 KB  relu(h1@W2+b2), col 63 = 0
    __shared__ float    sMask[LXN];        //  2 KB
    __shared__ float    sR[1024];          //  4 KB  per-column T/S ratios

    const int split = blockIdx.x;          // 0 or 1
    const int n     = blockIdx.y;
    const int tid   = threadIdx.x;
    const int lane  = tid & 31;
    const int wave  = tid >> 5;
    const int m16   = lane & 15;           // A row / B col / D col selector
    const int half  = lane >> 4;

    // ---------------- stage X[n] and mask into LDS ----------------
    const float* Xn = pX + (size_t)n * (LXN * D_IN);
    for (int i = tid; i < LXN * D_IN; i += 256) sX[i] = (_Float16)Xn[i];
    const float* Mn = pMask + (size_t)n * LXN;
    for (int i = tid; i < LXN; i += 256) sMask[i] = Mn[i];
    __syncthreads();

    // ---------------- layer 1: h1 = relu(X @ W1 + b1), K = 32 ----------------
    {
        v16h bw[4];
        for (int ct = 0; ct < 4; ++ct) {
            const int ncol = ct * 16 + m16;
            v16h b;
            #pragma unroll
            for (int i = 0; i < 16; ++i) {
                const int k = 8 * half + (i & 7) + 16 * (i >> 3); // 0..31
                b[i] = (_Float16)((ncol < K_DIM) ? pW1[k * K_DIM + ncol] : 0.f);
            }
            bw[ct] = b;
        }
        for (int rt = wave; rt < 32; rt += 8) {
            const int am = rt * 16 + m16;                        // A row
            v8h lo = *(const v8h*)&sX[am * D_IN +      8 * half];
            v8h hi = *(const v8h*)&sX[am * D_IN + 16 + 8 * half];
            v16h a = cat8(lo, hi);
            const int rowb = rt * 16 + 8 * half;                 // D rows base
            for (int ct = 0; ct < 4; ++ct) {
                v8f acc = {};
                acc = __builtin_amdgcn_wmma_f32_16x16x32_f16(
                        false, a, false, bw[ct], (short)0, acc, false, false);
                const int ncol = ct * 16 + m16;
                const float bv = (ncol < K_DIM) ? pB1[ncol] : 0.f;
                #pragma unroll
                for (int r = 0; r < 8; ++r) {
                    float h = acc[r] + bv;
                    sH1[(rowb + r) * HID + ncol] = (_Float16)(h > 0.f ? h : 0.f);
                }
            }
        }
    }
    __syncthreads();

    // ---------------- layer 2: h2 = relu(h1 @ W2 + b2), K = 64 (padded) -------
    {
        v16h bw[4][2];
        for (int ct = 0; ct < 4; ++ct) {
            const int ncol = ct * 16 + m16;
            #pragma unroll
            for (int q = 0; q < 2; ++q) {
                v16h b;
                #pragma unroll
                for (int i = 0; i < 16; ++i) {
                    const int k = 32 * q + 8 * half + (i & 7) + 16 * (i >> 3); // 0..63
                    b[i] = (_Float16)((k < K_DIM && ncol < K_DIM)
                                      ? pW2[k * K_DIM + ncol] : 0.f);
                }
                bw[ct][q] = b;
            }
        }
        for (int rt = wave; rt < 32; rt += 8) {
            const int am = rt * 16 + m16;
            v16h a0 = cat8(*(const v8h*)&sH1[am * HID +      8 * half],
                           *(const v8h*)&sH1[am * HID + 16 + 8 * half]);
            v16h a1 = cat8(*(const v8h*)&sH1[am * HID + 32 + 8 * half],
                           *(const v8h*)&sH1[am * HID + 48 + 8 * half]);
            const int rowb = rt * 16 + 8 * half;
            for (int ct = 0; ct < 4; ++ct) {
                v8f acc = {};
                acc = __builtin_amdgcn_wmma_f32_16x16x32_f16(
                        false, a0, false, bw[ct][0], (short)0, acc, false, false);
                acc = __builtin_amdgcn_wmma_f32_16x16x32_f16(
                        false, a1, false, bw[ct][1], (short)0, acc, false, false);
                const int ncol = ct * 16 + m16;
                const float bv = (ncol < K_DIM) ? pB2[ncol] : 0.f;
                #pragma unroll
                for (int r = 0; r < 8; ++r) {
                    float h = acc[r] + bv;
                    sH2[(rowb + r) * HID + ncol] = (_Float16)(h > 0.f ? h : 0.f);
                }
            }
        }
    }
    __syncthreads();

    // -------- layer 3 + online masked softmax over Lx + X-weighted pool -------
    // Column pair p covers cols [k*32, k*32+32) with k = split*32 + p, i.e.
    // exactly one output k per pair. Each wave streams 512 rows through WMMA
    // keeping flash-softmax stats (m, S, T) per column in registers.
    const int kbase  = split * 32;
    const int npairs = (split == 0) ? 32 : 31;
    for (int p = wave; p < npairs; p += 8) {
        const int k = kbase + p;
        #pragma unroll
        for (int t = 0; t < 2; ++t) {
            const int gcol = k * 32 + t * 16 + m16;              // global column
            // B fragments of W3 (K = 64, rows 63.. padded 0), reused 32x
            v16h b0, b1;
            #pragma unroll
            for (int i = 0; i < 16; ++i) {
                const int k0 = 8 * half + (i & 7) + 16 * (i >> 3);
                const int k1 = 32 + 8 * half + (i & 7) + 16 * (i >> 3);
                b0[i] = (_Float16)pW3[(size_t)k0 * COLS + gcol];
                b1[i] = (_Float16)((k1 < K_DIM) ? pW3[(size_t)k1 * COLS + gcol] : 0.f);
            }
            const float b3v = pB3[gcol];
            const int   dcol = gcol & 31;
            float mrun = NEG_BIG, S = 0.f, T = 0.f;
            for (int rt = 0; rt < 32; ++rt) {
                const int am = rt * 16 + m16;
                v16h a0 = cat8(*(const v8h*)&sH2[am * HID +      8 * half],
                               *(const v8h*)&sH2[am * HID + 16 + 8 * half]);
                v16h a1 = cat8(*(const v8h*)&sH2[am * HID + 32 + 8 * half],
                               *(const v8h*)&sH2[am * HID + 48 + 8 * half]);
                v8f acc = {};
                acc = __builtin_amdgcn_wmma_f32_16x16x32_f16(
                        false, a0, false, b0, (short)0, acc, false, false);
                acc = __builtin_amdgcn_wmma_f32_16x16x32_f16(
                        false, a1, false, b1, (short)0, acc, false, false);
                const int rowb = rt * 16 + 8 * half;             // D rows base
                float v[8]; float vm = NEG_BIG;
                #pragma unroll
                for (int r = 0; r < 8; ++r) {
                    float fv = acc[r] + b3v;
                    fv = (sMask[rowb + r] > 0.5f) ? fv : NEG_BIG; // masked -> -1e8
                    v[r] = fv;
                    vm = fmaxf(vm, fv);
                }
                const float mn    = fmaxf(mrun, vm);
                const float scale = __builtin_amdgcn_exp2f((mrun - mn) * L2E);
                float se = 0.f, te = 0.f;
                #pragma unroll
                for (int r = 0; r < 8; ++r) {
                    const float e = __builtin_amdgcn_exp2f((v[r] - mn) * L2E);
                    se += e;
                    te += e * (float)sX[(rowb + r) * D_IN + dcol];
                }
                S = S * scale + se;
                T = T * scale + te;
                mrun = mn;
            }
            // combine the two half-wave row partitions (lane <-> lane^16)
            const float m2 = __shfl_xor(mrun, 16, 32);
            const float S2 = __shfl_xor(S,    16, 32);
            const float T2 = __shfl_xor(T,    16, 32);
            const float mm  = fmaxf(mrun, m2);
            const float sc1 = __builtin_amdgcn_exp2f((mrun - mm) * L2E);
            const float sc2 = __builtin_amdgcn_exp2f((m2   - mm) * L2E);
            const float Sf  = S * sc1 + S2 * sc2;                // > 0 always
            const float Tf  = T * sc1 + T2 * sc2;
            if (half == 0) sR[p * 32 + t * 16 + m16] = Tf / Sf;
        }
    }
    __syncthreads();

    // ---------------- out[n,k] = relu(T_bias[k] + sum_d R[k,d]) ----------------
    const int nk = (split == 0) ? 32 : 31;
    if (tid < nk) {
        const int k = kbase + tid;
        float s = pTB[k];
        #pragma unroll
        for (int d = 0; d < 32; ++d) s += sR[tid * 32 + d];
        pOut[n * K_DIM + k] = s > 0.f ? s : 0.f;
    }
}

extern "C" void kernel_launch(void* const* d_in, const int* in_sizes, int n_in,
                              void* d_out, int out_size, void* d_ws, size_t ws_size,
                              hipStream_t stream) {
    (void)in_sizes; (void)n_in; (void)out_size; (void)d_ws; (void)ws_size;
    const float* X  = (const float*)d_in[0];
    const float* Mk = (const float*)d_in[1];
    const float* W1 = (const float*)d_in[2];
    const float* B1 = (const float*)d_in[3];
    const float* W2 = (const float*)d_in[4];
    const float* B2 = (const float*)d_in[5];
    const float* W3 = (const float*)d_in[6];
    const float* B3 = (const float*)d_in[7];
    const float* TB = (const float*)d_in[8];
    float* OUT = (float*)d_out;

    dim3 grid(2, NBATCH);   // (k-split, batch)
    dim3 block(256);        // 8 wave32s
    ttcn_fused<<<grid, block, 0, stream>>>(X, Mk, W1, B1, W2, B2, W3, B3, TB, OUT);
}
